// ScaledDotProductAttention_53549652246682
// MI455X (gfx1250) — matrix operand order repaired and checked by reading, not compiled, not run on hardware
//
#include <hip/hip_runtime.h>
#include <hip/hip_bf16.h>
#include <math.h>
#include <stdint.h>

typedef __attribute__((ext_vector_type(16))) _Float16 v16h;
typedef __attribute__((ext_vector_type(8)))  _Float16 h8;
typedef __attribute__((ext_vector_type(2)))  _Float16 h2;
typedef __attribute__((ext_vector_type(8)))  float    v8f;
typedef __attribute__((ext_vector_type(4)))  float    f32x4;

#define NKV    4096
#define DHEAD  64
#define DMODEL 512
#define KTILE  32
#define WAVES  4
#define KBYTES (KTILE * DHEAD * 4)   // one K tile in f32 = 8192 B

__device__ __forceinline__ v8f wmma_f16(v16h a, v16h b, v8f c) {
  // D = A(16x32 f16) * B(32x16 f16) + C(16x16 f32)
  return __builtin_amdgcn_wmma_f32_16x16x32_f16(
      /*neg_a=*/false, a, /*neg_b=*/false, b,
      /*c_mod=*/(short)0, c, /*reuse_a=*/false, /*reuse_b=*/false);
}

// 16 consecutive f32 (global or LDS) -> f16 B-fragment half.
__device__ __forceinline__ v16h load16_cvt(const float* __restrict__ p) {
  v16h r;
#pragma unroll
  for (int i = 0; i < 16; i += 4) {
    f32x4 x = *(const f32x4*)(p + i);
#pragma unroll
    for (int j = 0; j < 4; ++j) r[i + j] = (_Float16)x[j];
  }
  return r;
}

// A-fragment (16x32 f16) from a global f32 row; elem e -> kdim = kbase + 16*lh + e
__device__ __forceinline__ v16h loadA_frag(const float* __restrict__ rowp,
                                           int kbase, int lh, float scl) {
  v16h r;
  const float* p0 = rowp + kbase + 16 * lh;
  const float* p1 = rowp + kbase + 8 + 16 * lh;
#pragma unroll
  for (int i = 0; i < 8; i += 4) {
    f32x4 x = *(const f32x4*)(p0 + i);
#pragma unroll
    for (int j = 0; j < 4; ++j) r[i + j] = (_Float16)(x[j] * scl);
  }
#pragma unroll
  for (int i = 0; i < 8; i += 4) {
    f32x4 x = *(const f32x4*)(p1 + i);
#pragma unroll
    for (int j = 0; j < 4; ++j) r[8 + i + j] = (_Float16)(x[j] * scl);
  }
  return r;
}

// A-fragment from row-major f16 LDS tile.
__device__ __forceinline__ v16h ldsA_frag(const _Float16* base, int row,
                                          int stride, int kbase, int lh) {
  const _Float16* p = base + row * stride + kbase + 16 * lh;
  h8 a = *(const h8*)p;
  h8 b = *(const h8*)(p + 8);
  v16h r;
#pragma unroll
  for (int e = 0; e < 8; ++e) { r[e] = a[e]; r[8 + e] = b[e]; }
  return r;
}

// B-fragment from LDS stored [N][K]; elem e -> k = kbase + 16*lh + e (contiguous)
__device__ __forceinline__ v16h ldsB_frag(const _Float16* base, int ncol,
                                          int kstride, int kbase, int lh) {
  const _Float16* p = base + ncol * kstride + kbase + 16 * lh;
  h8 a = *(const h8*)p;
  h8 b = *(const h8*)(p + 8);
  v16h r;
#pragma unroll
  for (int e = 0; e < 8; ++e) { r[e] = a[e]; r[8 + e] = b[e]; }
  return r;
}

__device__ __forceinline__ float hmax16(float x) {
  x = fmaxf(x, __shfl_xor(x, 1, 32));
  x = fmaxf(x, __shfl_xor(x, 2, 32));
  x = fmaxf(x, __shfl_xor(x, 4, 32));
  x = fmaxf(x, __shfl_xor(x, 8, 32));
  return x;
}

// LDS byte offset of a generic pointer into LDS (ISA 10.2: LDS addr = addr[31:0])
__device__ __forceinline__ uint32_t lds_off(const void* p) {
  return (uint32_t)(uintptr_t)p;
}

__global__ __launch_bounds__(WAVES * 32)
void attn_fc_kernel(const float* __restrict__ q, const float* __restrict__ k,
                    const float* __restrict__ v, const float* __restrict__ fcw,
                    const float* __restrict__ fcb, float* __restrict__ out) {
  extern __shared__ __align__(16) char smem[];
  float*    Kbuf  = (float*)smem;                          // [2][KTILE][DHEAD] f32, async-staged
  _Float16* WldsT = (_Float16*)(smem + 2 * KBYTES);        // [DMODEL][DHEAD]  (W^T f16)
  _Float16* VldsT = WldsT + DMODEL * DHEAD;                // [DHEAD][KTILE]   (V^T f16)
  _Float16* Plds  = VldsT + DHEAD * KTILE;                 // [WAVES][16][32]
  _Float16* Olds  = Plds + WAVES * 16 * 32;                // [WAVES][16][64]

  const int tid  = threadIdx.x;
  const int wave = tid >> 5;
  const int lane = tid & 31;
  const int lh   = lane >> 4;
  const int ll   = lane & 15;

  const int blocksPerBatch = NKV / (16 * WAVES);           // 64
  const int batch = blockIdx.x / blocksPerBatch;
  const int qtile = (blockIdx.x % blocksPerBatch) * WAVES + wave;
  const int q0    = qtile * 16;

  const float* kbat = k + (size_t)batch * NKV * DHEAD;
  const float* vbat = v + (size_t)batch * NKV * DHEAD;

  // ---- Stage FC weights (f32 -> f16), transposed: WldsT[n][kd] ----
  for (int g = tid; g < DHEAD * DMODEL; g += blockDim.x) {
    int kd = g >> 9;
    int n  = g & (DMODEL - 1);
    WldsT[n * DHEAD + kd] = (_Float16)fcw[g];
  }

  // ---- Q A-fragments; fold scale * log2(e) so softmax uses raw exp2 ----
  const float qscale = 0.125f * 1.44269504088896340736f;
  const float* qrow = q + ((size_t)batch * NKV + q0 + ll) * DHEAD;
  v16h aq0 = loadA_frag(qrow, 0, lh, qscale);
  v16h aq1 = loadA_frag(qrow, 32, lh, qscale);

  v16h ones;
#pragma unroll
  for (int e = 0; e < 16; ++e) ones[e] = (_Float16)1.0f;

  float mrow[8];
  v8f oacc[5];  // [0..3] = O columns 0..63 ; [4] = row-sum (P @ ones)
#pragma unroll
  for (int r = 0; r < 8; ++r) mrow[r] = -INFINITY;
#pragma unroll
  for (int t = 0; t < 5; ++t) { v8f z = {}; oacc[t] = z; }

  _Float16* Pw = Plds + wave * 16 * 32;

  // ---- async prefetch of one K tile (f32, linear) into LDS double buffer ----
  // 512B per instruction per wave; 4 instructions/wave -> 8KB tile per block.
  const uint32_t kbufBase = lds_off(Kbuf);
  auto prefetchK = [&](int kbn, int buf) {
    uint32_t lo = kbufBase + (uint32_t)buf * KBYTES + (uint32_t)tid * 64u;
    uint64_t ga = (uint64_t)(uintptr_t)(kbat + (size_t)kbn * DHEAD) + (uint64_t)tid * 64u;
#pragma unroll
    for (int c = 0; c < 4; ++c) {
      asm volatile("global_load_async_to_lds_b128 %0, %1, off"
                   :: "v"(lo + c * 16u), "v"(ga + c * 16u)
                   : "memory");
    }
  };

  prefetchK(0, 0);
  __syncthreads();  // WldsT staged

  for (int kb = 0; kb < NKV; kb += KTILE) {
    const int cur = (kb >> 5) & 1;
    prefetchK((kb + KTILE) & (NKV - 1), cur ^ 1);  // wrap on tail: uniform wait bound

    // ---- cooperative V tile: global f32 -> LDS f16, transposed [dim][key] ----
    // Key pairs let f32->f16 use v_cvt_pk and LDS writes use ds_store_b32.
    {
      int kp = tid & 15;            // key pair -> keys 2kp, 2kp+1
      int dg = (tid >> 4) * 8;      // 8 dims
      const float* v0 = vbat + (size_t)(kb + 2 * kp) * DHEAD + dg;
      f32x4 a0 = *(const f32x4*)v0,          a1 = *(const f32x4*)(v0 + 4);
      f32x4 b0 = *(const f32x4*)(v0 + DHEAD), b1 = *(const f32x4*)(v0 + DHEAD + 4);
#pragma unroll
      for (int i = 0; i < 8; ++i) {
        float x = (i < 4) ? a0[i] : a1[i - 4];
        float y = (i < 4) ? b0[i] : b1[i - 4];
        h2 pk; pk[0] = (_Float16)x; pk[1] = (_Float16)y;
        *(h2*)&VldsT[(dg + i) * KTILE + 2 * kp] = pk;
      }
    }

    // oldest 4 async ops (current tile) complete; 4 newest may stay in flight
    asm volatile("s_wait_asynccnt 4" ::: "memory");
    __syncthreads();  // VldsT + Kbuf[cur] visible to all waves

    // ---- S = (Q*scale') K^T from LDS-staged K tile ----
    const float* Kc = Kbuf + cur * KTILE * DHEAD;
    v8f s0 = {}, s1 = {};
    {
      const float* kr0 = Kc + (0 + ll) * DHEAD + 16 * lh;
      s0 = wmma_f16(aq0, load16_cvt(kr0), s0);
      s0 = wmma_f16(aq1, load16_cvt(kr0 + 32), s0);
      const float* kr1 = Kc + (16 + ll) * DHEAD + 16 * lh;
      s1 = wmma_f16(aq0, load16_cvt(kr1), s1);
      s1 = wmma_f16(aq1, load16_cvt(kr1 + 32), s1);
    }

    // ---- online softmax (base-2 domain); row sums come from the ones-WMMA ----
#pragma unroll
    for (int r = 0; r < 8; ++r) {
      float mx   = hmax16(fmaxf(s0[r], s1[r]));
      float mnew = fmaxf(mrow[r], mx);
      float corr = exp2f(mrow[r] - mnew);
      float p0   = exp2f(s0[r] - mnew);
      float p1   = exp2f(s1[r] - mnew);
      mrow[r] = mnew;
#pragma unroll
      for (int t = 0; t < 5; ++t) oacc[t][r] *= corr;
      int prow = r + 8 * lh;
      Pw[prow * 32 + ll]      = (_Float16)p0;
      Pw[prow * 32 + 16 + ll] = (_Float16)p1;
    }

    // ---- O += P(16x32) V(32x64);  l += P @ ones ----
    v16h pa = ldsA_frag(Pw, ll, 32, 0, lh);
#pragma unroll
    for (int t = 0; t < 4; ++t) {
      v16h bv = ldsB_frag(VldsT, 16 * t + ll, KTILE, 0, lh);
      oacc[t] = wmma_f16(pa, bv, oacc[t]);
    }
    oacc[4] = wmma_f16(pa, ones, oacc[4]);

    __syncthreads();  // protect VldsT (and Kbuf[cur] WAR) before next iteration
  }

  // ---- normalize + stage O (f16 row-major [16][64]) for FC A-fragments ----
  _Float16* Ow = Olds + wave * 16 * 64;
#pragma unroll
  for (int r = 0; r < 8; ++r) {
    float inv = 1.0f / oacc[4][r];   // row sum (identical across the 16-lane half)
    int orow = r + 8 * lh;
#pragma unroll
    for (int t = 0; t < 4; ++t)
      Ow[orow * 64 + 16 * t + ll] = (_Float16)(oacc[t][r] * inv);
  }
  __syncthreads();

  v16h ao0 = ldsA_frag(Ow, ll, 64, 0, lh);
  v16h ao1 = ldsA_frag(Ow, ll, 64, 32, lh);

  // ---- FC: (16x64) @ (64x512) + bias, bias injected via the C operand ----
  float* obase = out + ((size_t)batch * NKV + q0) * DMODEL;
  for (int n = 0; n < DMODEL / 16; ++n) {
    float bias = fcb[n * 16 + ll];
    v8f c;
#pragma unroll
    for (int r = 0; r < 8; ++r) c[r] = bias;
    v16h bw0 = ldsB_frag(WldsT, n * 16 + ll, DHEAD, 0, lh);
    v16h bw1 = ldsB_frag(WldsT, n * 16 + ll, DHEAD, 32, lh);
    v8f d = wmma_f16(ao0, bw0, c);
    d = wmma_f16(ao1, bw1, d);
#pragma unroll
    for (int r = 0; r < 8; ++r)
      obase[(size_t)(r + 8 * lh) * DMODEL + n * 16 + ll] = d[r];
  }
}

extern "C" void kernel_launch(void* const* d_in, const int* in_sizes, int n_in,
                              void* d_out, int out_size, void* d_ws, size_t ws_size,
                              hipStream_t stream) {
  const float* q   = (const float*)d_in[0];
  const float* k   = (const float*)d_in[1];
  const float* v   = (const float*)d_in[2];
  const float* fcw = (const float*)d_in[3];
  const float* fcb = (const float*)d_in[4];
  float* out = (float*)d_out;

  int batches = in_sizes[0] / (NKV * DHEAD);               // 8
  int blocksPerBatch = NKV / (16 * WAVES);                 // 64
  dim3 grid(batches * blocksPerBatch), block(WAVES * 32);
  size_t smem = 2 * KBYTES +
                (size_t)(DMODEL * DHEAD + DHEAD * KTILE +
                         WAVES * 16 * 32 + WAVES * 16 * 64) * sizeof(_Float16);
  attn_fc_kernel<<<grid, block, smem, stream>>>(q, k, v, fcw, fcb, out);
}
